// Network_72859825209609
// MI455X (gfx1250) — matrix operand order repaired and checked by reading, not compiled, not run on hardware
//
#include <hip/hip_runtime.h>
#include <math.h>

typedef __attribute__((ext_vector_type(16))) _Float16 v16h;
typedef __attribute__((ext_vector_type(8)))  float    v8f;

#define TM 64
#define TN 128
#define TKK 32
#define LDP 48   // LDS row pitch in halfs (96B rows, 16B aligned)

#if defined(__has_builtin)
#  if __has_builtin(__builtin_amdgcn_sched_barrier)
#    define SCHED_FENCE() __builtin_amdgcn_sched_barrier(0)
#  endif
#endif
#ifndef SCHED_FENCE
#  define SCHED_FENCE() asm volatile("" ::: "memory")
#endif

// ---------------------------------------------------------------------------
// Stage 1: global -> registers (issue loads, no conversion, no LDS).
// KG: guard/zero-pad K beyond Kd. NG: guard/zero-pad N beyond Nc. M full.
// ---------------------------------------------------------------------------
template<bool TRANSB, bool KG, bool NG>
__device__ __forceinline__ void fetch_tile(
    const float* __restrict__ A, const float* __restrict__ B,
    float (&ra)[8], float (&rb)[16],
    int m0, int n0, int Nc, int Kd, int k0, int tid)
{
    // ---- A tile (64 x 32): 2 x float4 per thread ----
    {
        const int r  = tid >> 3;        // 0..31
        const int c4 = (tid & 7) * 4;   // 0..28
        #pragma unroll
        for (int rr = 0; rr < 2; rr++) {
            const float* p = A + (size_t)(m0 + r + rr * 32) * Kd;
            if (!KG) {
                const float4 q = *(const float4*)(p + k0 + c4);
                ra[rr*4+0] = q.x; ra[rr*4+1] = q.y; ra[rr*4+2] = q.z; ra[rr*4+3] = q.w;
            } else {
                #pragma unroll
                for (int i = 0; i < 4; i++) {
                    const int k  = k0 + c4 + i;
                    const int kc = k < Kd ? k : Kd - 1;
                    const float t = p[kc];
                    ra[rr*4+i] = k < Kd ? t : 0.f;
                }
            }
        }
    }
    if (!TRANSB) {
        // B[Kd,Nc] row-major: 4 x float4 along n (coalesced)
        const int c4n = (tid & 31) * 4;   // 0..124
        const int kb  = tid >> 5;         // 0..7
        #pragma unroll
        for (int j = 0; j < 4; j++) {
            const int k = k0 + kb + j * 8;
            if (!KG && !NG) {
                const float4 q = *(const float4*)(B + (size_t)k * Nc + n0 + c4n);
                rb[j*4+0] = q.x; rb[j*4+1] = q.y; rb[j*4+2] = q.z; rb[j*4+3] = q.w;
            } else {
                const int kc = k < Kd ? k : Kd - 1;
                #pragma unroll
                for (int i = 0; i < 4; i++) {
                    const int n   = n0 + c4n + i;
                    const int nc2 = (!NG || n < Nc) ? n : Nc - 1;
                    const float t = B[(size_t)kc * Nc + nc2];
                    rb[j*4+i] = (k < Kd && (!NG || n < Nc)) ? t : 0.f;
                }
            }
        }
    } else {
        // B^T given as Bt[Nc,Kd] row-major: 4 x float4 along k
        #pragma unroll
        for (int j = 0; j < 4; j++) {
            const int n  = j * 32 + (tid >> 3);
            const int c4 = (tid & 7) * 4;
            const bool nok = !NG || (n0 + n) < Nc;
            const int  nr  = nok ? (n0 + n) : Nc - 1;
            const float* p = B + (size_t)nr * Kd;
            if (!KG && !NG) {
                const float4 q = *(const float4*)(p + k0 + c4);
                rb[j*4+0] = q.x; rb[j*4+1] = q.y; rb[j*4+2] = q.z; rb[j*4+3] = q.w;
            } else {
                #pragma unroll
                for (int i = 0; i < 4; i++) {
                    const int k  = k0 + c4 + i;
                    const int kc = k < Kd ? k : Kd - 1;
                    const float t = p[kc];
                    rb[j*4+i] = (k < Kd && nok) ? t : 0.f;
                }
            }
        }
    }
}

// ---------------------------------------------------------------------------
// Stage 2: registers -> LDS (waits for loads, converts f32->f16, stores).
// B is stored K-transposed: Bs[n][k].
// ---------------------------------------------------------------------------
template<bool TRANSB>
__device__ __forceinline__ void store_tile_lds(
    const float (&ra)[8], const float (&rb)[16],
    _Float16* __restrict__ As, _Float16* __restrict__ Bs, int tid)
{
    const int r  = tid >> 3;
    const int c4 = (tid & 7) * 4;
    #pragma unroll
    for (int rr = 0; rr < 2; rr++) {
        _Float16* dst = &As[(r + rr * 32) * LDP + c4];
        #pragma unroll
        for (int i = 0; i < 4; i++) dst[i] = (_Float16)ra[rr*4+i];
    }
    if (!TRANSB) {
        const int c4n = (tid & 31) * 4;
        const int kb  = tid >> 5;
        #pragma unroll
        for (int j = 0; j < 4; j++) {
            const int k = kb + j * 8;
            #pragma unroll
            for (int i = 0; i < 4; i++)
                Bs[(c4n + i) * LDP + k] = (_Float16)rb[j*4+i];
        }
    } else {
        #pragma unroll
        for (int j = 0; j < 4; j++) {
            const int n = j * 32 + (tid >> 3);
            _Float16* dst = &Bs[n * LDP + c4];
            #pragma unroll
            for (int i = 0; i < 4; i++) dst[i] = (_Float16)rb[j*4+i];
        }
    }
}

// 2x2 WMMA wave tile from LDS fragments
__device__ __forceinline__ void wmma_compute(const _Float16* As, const _Float16* Bs,
                                             v8f (&acc)[2][2], int wm, int wn, int hv, int l16)
{
    v16h afrag[2], bfrag[2];
    #pragma unroll
    for (int sm = 0; sm < 2; sm++) {
        const int m = wm * 32 + sm * 16 + l16;
        const _Float16* ap  = &As[m * LDP + hv * 16];
        #pragma unroll
        for (int i = 0; i < 16; i++) afrag[sm][i] = ap[i];
    }
    #pragma unroll
    for (int sn = 0; sn < 2; sn++) {
        const int n = wn * 32 + sn * 16 + l16;
        const _Float16* bp = &Bs[n * LDP + hv * 16];
        #pragma unroll
        for (int i = 0; i < 16; i++) bfrag[sn][i] = bp[i];
    }
    #pragma unroll
    for (int sm = 0; sm < 2; sm++)
        #pragma unroll
        for (int sn = 0; sn < 2; sn++)
            acc[sm][sn] = __builtin_amdgcn_wmma_f32_16x16x32_f16(
                false, afrag[sm], false, bfrag[sn], (short)0, acc[sm][sn], false, false);
}

__device__ __forceinline__ float apply_epi(float val, const float* bias, int n, int act)
{
    if (bias) val += bias[n];
    if (act == 1)      val = tanhf(val);
    else if (act == 2) val = 1.f / (1.f + __expf(-val));
    return val;
}

// ---------------------------------------------------------------------------
// Fast GEMM: full 64x128 tile. Double-buffered LDS. Scheduling fences pin the
// order: issue global loads -> WMMAs (overlap) -> wait/convert/LDS-store.
// ---------------------------------------------------------------------------
template<bool TRANSB>
__global__ __launch_bounds__(256) void wmma_gemm_fast(
    const float* __restrict__ A, const float* __restrict__ B,
    float* __restrict__ C, int M, int Nc, int Kd,
    const float* __restrict__ bias, int act)
{
    __shared__ _Float16 As[2][TM * LDP];
    __shared__ _Float16 Bs[2][TN * LDP];

    const int tid  = threadIdx.x;
    const int lane = tid & 31;
    const int wave = tid >> 5;
    const int wm   = wave >> 2;
    const int wn   = wave & 3;
    const int hv   = lane >> 4;
    const int l16  = lane & 15;
    const int m0   = blockIdx.y * TM;
    const int n0   = blockIdx.x * TN;

    v8f acc[2][2] = {};
    const int  Tfull = Kd / TKK;          // >= 1 for all shapes here (Kd >= 512)
    const bool rem   = (Kd % TKK) != 0;

    float ra[8], rb[16];
    fetch_tile<TRANSB, false, false>(A, B, ra, rb, m0, n0, Nc, Kd, 0, tid);
    store_tile_lds<TRANSB>(ra, rb, As[0], Bs[0], tid);
    __syncthreads();

    int buf = 0;
    for (int t = 0; t + 1 < Tfull; t++) {
        // issue next tile's global loads (no wait yet)
        fetch_tile<TRANSB, false, false>(A, B, ra, rb, m0, n0, Nc, Kd, (t + 1) * TKK, tid);
        if ((t + 2) * TKK < Kd) {           // prefetch tile t+2 (global_prefetch_b8)
            const int k2 = (t + 2) * TKK;
            __builtin_prefetch(A + (size_t)(m0 + (tid >> 3)) * Kd + k2, 0, 1);
            if (!TRANSB) __builtin_prefetch(B + (size_t)(k2 + (tid >> 5)) * Nc + n0 + (tid & 31) * 4, 0, 1);
            else         __builtin_prefetch(B + (size_t)(n0 + (tid >> 3)) * Kd + k2, 0, 1);
        }
        SCHED_FENCE();   // keep load-issue above the matrix block
        // matrix math overlaps the in-flight loads
        wmma_compute(As[buf], Bs[buf], acc, wm, wn, hv, l16);
        SCHED_FENCE();   // forbid hoisting the wait/convert/store above the WMMAs
        // wait + convert + LDS store into the other buffer
        store_tile_lds<TRANSB>(ra, rb, As[buf ^ 1], Bs[buf ^ 1], tid);
        __syncthreads();
        buf ^= 1;
    }
    if (rem) {
        fetch_tile<TRANSB, true, false>(A, B, ra, rb, m0, n0, Nc, Kd, Tfull * TKK, tid);
        SCHED_FENCE();
        wmma_compute(As[buf], Bs[buf], acc, wm, wn, hv, l16);
        SCHED_FENCE();
        store_tile_lds<TRANSB>(ra, rb, As[buf ^ 1], Bs[buf ^ 1], tid);
        __syncthreads();
        wmma_compute(As[buf ^ 1], Bs[buf ^ 1], acc, wm, wn, hv, l16);
    } else {
        wmma_compute(As[buf], Bs[buf], acc, wm, wn, hv, l16);
    }

    #pragma unroll
    for (int sm = 0; sm < 2; sm++)
        #pragma unroll
        for (int sn = 0; sn < 2; sn++)
            #pragma unroll
            for (int v = 0; v < 8; v++) {
                const int m = m0 + wm * 32 + sm * 16 + hv * 8 + v;
                const int n = n0 + wn * 32 + sn * 16 + l16;
                C[(size_t)m * Nc + n] = apply_epi(acc[sm][sn][v], bias, n, act);
            }
}

// ---------------------------------------------------------------------------
// Edge GEMM: Nc%128 strip (Nc=10 / 1800 tails). Branchless clamped guards.
// ---------------------------------------------------------------------------
template<bool TRANSB>
__global__ __launch_bounds__(256) void wmma_gemm_edge(
    const float* __restrict__ A, const float* __restrict__ B,
    float* __restrict__ C, int M, int Nc, int Kd,
    const float* __restrict__ bias, int act, int n_origin)
{
    __shared__ _Float16 As[TM * LDP];
    __shared__ _Float16 Bs[TN * LDP];

    const int tid  = threadIdx.x;
    const int lane = tid & 31;
    const int wave = tid >> 5;
    const int wm   = wave >> 2;
    const int wn   = wave & 3;
    const int hv   = lane >> 4;
    const int l16  = lane & 15;
    const int m0   = blockIdx.y * TM;
    const int n0   = n_origin + blockIdx.x * TN;

    v8f acc[2][2] = {};
    float ra[8], rb[16];

    for (int k0 = 0; k0 < Kd; k0 += TKK) {
        fetch_tile<TRANSB, true, true>(A, B, ra, rb, m0, n0, Nc, Kd, k0, tid);
        store_tile_lds<TRANSB>(ra, rb, As, Bs, tid);
        __syncthreads();
        wmma_compute(As, Bs, acc, wm, wn, hv, l16);
        __syncthreads();
    }

    #pragma unroll
    for (int sm = 0; sm < 2; sm++)
        #pragma unroll
        for (int sn = 0; sn < 2; sn++)
            #pragma unroll
            for (int v = 0; v < 8; v++) {
                const int m = m0 + wm * 32 + sm * 16 + hv * 8 + v;
                const int n = n0 + wn * 32 + sn * 16 + l16;
                if (n < Nc)
                    C[(size_t)m * Nc + n] = apply_epi(acc[sm][sn][v], bias, n, act);
            }
}

// ---------------------------------------------------------------------------
__global__ void zero_f32(float* p, size_t n) {
    size_t stride = (size_t)gridDim.x * blockDim.x;
    for (size_t i = (size_t)blockIdx.x * blockDim.x + threadIdx.x; i < n; i += stride)
        p[i] = 0.f;
}

// Top-K smallest distances per row from Gram matrix; symmetric scatter of 1.0
__global__ void topk_scatter(const float* __restrict__ gram, float* __restrict__ g,
                             int Nn, int Kk)
{
    __shared__ int   sel[16];
    __shared__ float sval[256];
    __shared__ int   sidx[256];
    const int i = blockIdx.x;
    const int t = threadIdx.x;
    const float sqi = gram[(size_t)i * Nn + i];

    for (int p = 0; p < Kk; p++) {
        float best = 3.4e38f;
        int   bidx = -1;
        for (int j = t; j < Nn; j += 256) {
            bool skip = false;
            for (int q = 0; q < p; q++) if (sel[q] == j) skip = true;
            if (skip) continue;
            const float d = sqi + gram[(size_t)j * Nn + j]
                          - 2.f * gram[(size_t)i * Nn + j];
            if (d < best) { best = d; bidx = j; }
        }
        sval[t] = best; sidx[t] = bidx;
        __syncthreads();
        for (int s = 128; s > 0; s >>= 1) {
            if (t < s && sval[t + s] < sval[t]) {
                sval[t] = sval[t + s]; sidx[t] = sidx[t + s];
            }
            __syncthreads();
        }
        if (t == 0) sel[p] = sidx[0];
        __syncthreads();
    }
    if (t < Kk) {
        const int j = sel[t];
        if (j >= 0 && j != i) {
            g[(size_t)i * Nn + j] = 1.f;
            g[(size_t)j * Nn + i] = 1.f;
        }
    }
}

__global__ void degree_rsqrt(const float* __restrict__ g, float* __restrict__ dinv, int Nn)
{
    __shared__ float red[256];
    const int i = blockIdx.x, t = threadIdx.x;
    float s = 0.f;
    for (int j = t; j < Nn; j += 256) s += g[(size_t)i * Nn + j];
    red[t] = s; __syncthreads();
    for (int st = 128; st > 0; st >>= 1) {
        if (t < st) red[t] += red[t + st];
        __syncthreads();
    }
    if (t == 0) dinv[i] = rsqrtf(red[0] + 1.0f);
}

__global__ void build_s(const float* __restrict__ g, const float* __restrict__ dinv,
                        float* __restrict__ S, int Nn)
{
    const size_t total = (size_t)Nn * Nn;
    const size_t stride = (size_t)gridDim.x * blockDim.x;
    for (size_t idx = (size_t)blockIdx.x * blockDim.x + threadIdx.x; idx < total; idx += stride) {
        const int i = (int)(idx / Nn), j = (int)(idx % Nn);
        const float v = g[idx] + (i == j ? 1.f : 0.f);
        S[idx] = dinv[i] * v * dinv[j];
    }
}

__global__ void l2norm_rows(const float* __restrict__ in, float* __restrict__ out, int Cc)
{
    __shared__ float red[128];
    const int i = blockIdx.x, t = threadIdx.x;
    const float v = (t < Cc) ? in[(size_t)i * Cc + t] : 0.f;
    red[t] = v * v; __syncthreads();
    for (int s = 64; s > 0; s >>= 1) {
        if (t < s) red[t] += red[t + s];
        __syncthreads();
    }
    const float inv = rsqrtf(red[0]);
    if (t < Cc) out[(size_t)i * Cc + t] = v * inv;
}

__global__ void softmax_rows(const float* __restrict__ in, float* __restrict__ out,
                             int Nr, int Cc)
{
    const int i = blockIdx.x * blockDim.x + threadIdx.x;
    if (i >= Nr) return;
    const float* r = in + (size_t)i * Cc;
    float mx = r[0];
    for (int c = 1; c < Cc; c++) mx = fmaxf(mx, r[c]);
    float e[16], s = 0.f;
    for (int c = 0; c < Cc; c++) { e[c] = __expf(r[c] - mx); s += e[c]; }
    const float inv = 1.f / s;
    for (int c = 0; c < Cc; c++) out[(size_t)i * Cc + c] = e[c] * inv;
}

// ---------------------------------------------------------------------------
extern "C" void kernel_launch(void* const* d_in, const int* in_sizes, int n_in,
                              void* d_out, int out_size, void* d_ws, size_t ws_size,
                              hipStream_t stream)
{
    (void)in_sizes; (void)n_in; (void)out_size; (void)ws_size;
    const int V = 2, N = 4096, D = 1024, HID = 1800, F = 512, HF = 128, C = 10, Kk = 10;

    const float* xs     = (const float*)d_in[0];
    const float* enc_w1 = (const float*)d_in[1];
    const float* enc_b1 = (const float*)d_in[2];
    const float* enc_w2 = (const float*)d_in[3];
    const float* enc_b2 = (const float*)d_in[4];
    const float* enc_W  = (const float*)d_in[5];
    const float* dec_W1 = (const float*)d_in[6];
    const float* dec_w1 = (const float*)d_in[7];
    const float* dec_b1 = (const float*)d_in[8];
    const float* dec_w2 = (const float*)d_in[9];
    const float* dec_b2 = (const float*)d_in[10];
    const float* fc_w   = (const float*)d_in[11];
    const float* fc_b   = (const float*)d_in[12];
    const float* lc_w   = (const float*)d_in[13];
    const float* lc_b   = (const float*)d_in[14];

    float* out = (float*)d_out;
    const size_t hs_off  = 0;
    const size_t qs_off  = hs_off  + (size_t)V * N * HF;
    const size_t xrs_off = qs_off  + (size_t)V * N * C;
    const size_t zs_off  = xrs_off + (size_t)V * N * D;
    const size_t gs_off  = zs_off  + (size_t)V * N * F;
    const size_t ars_off = gs_off  + (size_t)V * N * N;

    float* wsf   = (float*)d_ws;
    float* S     = wsf;                          // N*N
    float* poolA = S     + (size_t)N * N;        // N*HID
    float* poolB = poolA + (size_t)N * HID;      // N*HID
    float* poolC = poolB + (size_t)N * HID;      // N*D
    float* hpre  = poolC + (size_t)N * D;        // N*HF
    float* qpre  = hpre  + (size_t)N * HF;       // N*C
    float* dinv  = qpre  + (size_t)N * C;        // N

    auto gemm = [&](const float* A, const float* B, float* Cc,
                    int M, int Nc, int Kd, const float* bias, int transB, int act) {
        const int nfull = Nc / TN;
        if (nfull > 0) {
            dim3 grid(nfull, M / TM);
            if (transB) wmma_gemm_fast<true ><<<grid, 256, 0, stream>>>(A, B, Cc, M, Nc, Kd, bias, act);
            else        wmma_gemm_fast<false><<<grid, 256, 0, stream>>>(A, B, Cc, M, Nc, Kd, bias, act);
        }
        if (Nc % TN) {
            dim3 grid(1, M / TM);
            if (transB) wmma_gemm_edge<true ><<<grid, 256, 0, stream>>>(A, B, Cc, M, Nc, Kd, bias, act, nfull * TN);
            else        wmma_gemm_edge<false><<<grid, 256, 0, stream>>>(A, B, Cc, M, Nc, Kd, bias, act, nfull * TN);
        }
    };

    for (int v = 0; v < V; v++) {
        const float* x   = xs     + (size_t)v * N * D;
        const float* ew1 = enc_w1 + (size_t)v * D * HID;
        const float* eb1 = enc_b1 + (size_t)v * HID;
        const float* ew2 = enc_w2 + (size_t)v * HID * F;
        const float* eb2 = enc_b2 + (size_t)v * F;
        const float* eWv = enc_W  + (size_t)v * F * F;
        const float* dW1 = dec_W1 + (size_t)v * F * F;
        const float* dw1 = dec_w1 + (size_t)v * F * HID;
        const float* db1 = dec_b1 + (size_t)v * HID;
        const float* dw2 = dec_w2 + (size_t)v * HID * D;
        const float* db2 = dec_b2 + (size_t)v * D;

        float* hs_v  = out + hs_off  + (size_t)v * N * HF;
        float* qs_v  = out + qs_off  + (size_t)v * N * C;
        float* xrs_v = out + xrs_off + (size_t)v * N * D;
        float* zs_v  = out + zs_off  + (size_t)v * N * F;
        float* gs_v  = out + gs_off  + (size_t)v * N * N;
        float* ars_v = out + ars_off + (size_t)v * N * N;

        // --- kNN graph + normalized adjacency ---
        gemm(x, x, ars_v, N, N, D, nullptr, 1, 0);            // Gram = x @ x^T (staged in ars)
        zero_f32<<<4096, 256, 0, stream>>>(gs_v, (size_t)N * N);
        topk_scatter<<<N, 256, 0, stream>>>(ars_v, gs_v, N, Kk);
        degree_rsqrt<<<N, 256, 0, stream>>>(gs_v, dinv, N);
        build_s<<<4096, 256, 0, stream>>>(gs_v, dinv, S, N);

        // --- encoder ---
        gemm(S, x, poolC, N, D, N, nullptr, 0, 0);            // sx = S @ x
        gemm(poolC, ew1, poolA, N, HID, D, eb1, 0, 1);        // o1 = tanh(sx@ew1+b)
        gemm(S, poolA, poolB, N, HID, N, nullptr, 0, 0);      // so1 = S @ o1
        gemm(poolB, ew2, zs_v, N, F, HID, eb2, 0, 1);         // z = tanh(so1@ew2+b)

        // --- adjacency reconstruction ---
        gemm(zs_v, eWv, poolC, N, F, F, nullptr, 0, 0);       // zW = z @ eW
        gemm(poolC, zs_v, ars_v, N, N, F, nullptr, 1, 2);     // a = sigmoid(zW @ z^T)

        // --- contrastive heads ---
        gemm(zs_v, fc_w, hpre, N, HF, F, fc_b, 0, 0);         // h_pre
        l2norm_rows<<<N, 128, 0, stream>>>(hpre, hs_v, HF);
        gemm(zs_v, lc_w, qpre, N, C, F, lc_b, 0, 0);          // q_pre
        softmax_rows<<<(N + 255) / 256, 256, 0, stream>>>(qpre, qs_v, N, C);

        // --- decoder ---
        gemm(zs_v, dW1, poolC, N, F, F, nullptr, 0, 1);                     // h1
        gemm(S, poolC, poolC + (size_t)N * F, N, F, N, nullptr, 0, 0);      // sh1
        gemm(poolC + (size_t)N * F, dw1, poolA, N, HID, F, db1, 0, 1);      // h11
        gemm(S, poolA, poolB, N, HID, N, nullptr, 0, 0);                    // sh11
        gemm(poolB, dw2, xrs_v, N, D, HID, db2, 0, 1);                      // xr
    }
}
